// LstmTl_21466246545969
// MI455X (gfx1250) — matrix operand order repaired and checked by reading, not compiled, not run on hardware
//
#include <hip/hip_runtime.h>

typedef __attribute__((ext_vector_type(16))) _Float16 v16h;
typedef __attribute__((ext_vector_type(8)))  _Float16 h8;
typedef __attribute__((ext_vector_type(8)))  float    v8f;

union V16U { v16h v; h8 h[2]; };

// A-matrix fragment (16x32 f16): lane = (hsel, idx); per ISA, lane holds row M=idx,
// K runs [hsel*8, +8) and [16+hsel*8, +8)  -> two 16B contiguous LDS loads.
__device__ __forceinline__ v16h load_fragA(const _Float16* base, int row, int stride,
                                           int kBase, int hsel) {
  V16U u;
  const _Float16* p = base + row * stride + kBase + hsel * 8;
  u.h[0] = *(const h8*)(p);
  u.h[1] = *(const h8*)(p + 16);
  return u.v;
}

// B-matrix fragment (32x16 f16) loaded from a transposed (column-major) LDS image
// Bt[n][k]: lane holds column N=idx, K runs [hsel*8, +8) and [16+hsel*8, +8).
__device__ __forceinline__ v16h load_fragB(const _Float16* base, int row, int stride,
                                           int kBase, int hsel) {
  V16U u;
  const _Float16* p = base + row * stride + kBase + hsel * 8;
  u.h[0] = *(const h8*)(p);
  u.h[1] = *(const h8*)(p + 16);
  return u.v;
}

__device__ __forceinline__ v8f wmma_f16(v16h a, v16h b, v8f c) {
  return __builtin_amdgcn_wmma_f32_16x16x32_f16(false, a, false, b, (short)0, c,
                                                false, false);
}

__device__ __forceinline__ float sigmoidf(float x) {
  return 1.0f / (1.0f + __expf(-x));
}

// LDS layout (halves). Row strides padded to multiples of 8 halves (16B) to keep
// 128-bit ds loads aligned while rotating banks.
#define SW1  72     // Ww1/Wu1 row stride (64 + pad)
#define SW2  136    // fused [Ww2|Wu2] row stride (128 + pad)
#define SXT  72     // Xt / Ht row stride
#define SC1  136    // fused [C1x|C1h] row stride

#define OFF_WW1 0
#define OFF_WU1 13824     // 3*64*72
#define OFF_W2  27648
#define OFF_XT  53760     // + 3*64*136
#define OFF_HT  58368     // + 64*72
#define OFF_C1  62976     // + 64*72
#define SMEM_HALVES 71680 // + 64*136  -> 143360 bytes

__global__ __launch_bounds__(512)
void lstm_tl_kernel(const float* __restrict__ inputs,  // (N=64, L=128, 64, 64)
                    const float* __restrict__ Ww1,     // (4, 64, 64), gates 0..2 used
                    const float* __restrict__ Ww2,
                    const float* __restrict__ Wu1,
                    const float* __restrict__ Wu2,
                    float* __restrict__ out)           // outs | h_last | c_last
{
  extern __shared__ _Float16 smem[];
  _Float16* sWw1 = smem + OFF_WW1;  // [g][a][i]
  _Float16* sWu1 = smem + OFF_WU1;  // [g][a][i]
  _Float16* sW2  = smem + OFF_W2;   // [g][b][k] : k<64 -> Ww2[g][b][k], else Wu2[g][b][k-64]
  _Float16* sXt  = smem + OFF_XT;   // [j][i] = X[i][j]   (transposed image for B-frags)
  _Float16* sHt  = smem + OFF_HT;   // [j][i] = h[i][j]
  _Float16* sC1  = smem + OFF_C1;   // [a][k] : k<64 -> (Ww1 X), else (Wu1 h)

  const int n    = blockIdx.x;
  const int tid  = threadIdx.x;
  const int wave = tid >> 5;        // wave32: 16 waves per block
  const int lane = tid & 31;
  const int hsel = lane >> 4;
  const int idx  = lane & 15;
  const int ti   = wave >> 2;       // output tile row  (16x16 tile in 64x64)
  const int tj   = wave & 3;        // output tile col

  // ---- Stage weights into LDS (f32 -> f16), fuse Ww2|Wu2 along K ----
  for (int e = tid; e < 3 * 64 * 64; e += 512) {
    int g = e >> 12, rest = e & 4095, a = rest >> 6, i = rest & 63;
    sWw1[(g * 64 + a) * SW1 + i]      = (_Float16)Ww1[e];
    sWu1[(g * 64 + a) * SW1 + i]      = (_Float16)Wu1[e];
    sW2 [(g * 64 + a) * SW2 + i]      = (_Float16)Ww2[e];
    sW2 [(g * 64 + a) * SW2 + 64 + i] = (_Float16)Wu2[e];
  }
  for (int e = tid; e < 64 * SXT; e += 512) sHt[e] = (_Float16)0.0f;

  // Persistent per-wave state fragments in WMMA C-layout:
  // element r of lane -> row = ti*16 + hsel*8 + r, col = tj*16 + idx
  v8f c_acc = {};
  v8f h_acc = {};
  v8f gate[3];

  const float* xbase = inputs + (size_t)n * 128 * 4096;
  float*       outs  = out    + (size_t)n * 128 * 4096;

  for (int t = 0; t < 128; ++t) {
    const float* X = xbase + (size_t)t * 4096;
    // Transposed f16 image of X_t
    for (int e = tid; e < 4096; e += 512) {
      int i = e >> 6, j = e & 63;
      sXt[j * SXT + i] = (_Float16)X[e];
    }
    if (t + 1 < 128)
      __builtin_prefetch(xbase + (size_t)(t + 1) * 4096 + tid * 8, 0, 0);
    __syncthreads();  // Xt ready; previous step's Ht writes published

    for (int g = 0; g < 3; ++g) {
      // Stage 1: C1x = Ww1[g] @ X,  C1h = Wu1[g] @ h   (this wave's 16x16 tiles)
      v8f accx = {}, acch = {};
      const _Float16* A1 = sWw1 + g * 64 * SW1;
      const _Float16* Ah = sWu1 + g * 64 * SW1;
      for (int kc = 0; kc < 64; kc += 32) {
        v16h aw = load_fragA(A1,  ti * 16 + idx, SW1, kc, hsel);
        v16h bx = load_fragB(sXt, tj * 16 + idx, SXT, kc, hsel);
        accx = wmma_f16(aw, bx, accx);
        v16h au = load_fragA(Ah,  ti * 16 + idx, SW1, kc, hsel);
        v16h bh = load_fragB(sHt, tj * 16 + idx, SXT, kc, hsel);
        acch = wmma_f16(au, bh, acch);
      }
      // Pack both results into fused 64x128 C1 (f16)
      {
        int row0 = ti * 16 + hsel * 8;
        int col  = tj * 16 + idx;
        for (int r = 0; r < 8; ++r) {
          sC1[(row0 + r) * SC1 + col]      = (_Float16)accx[r];
          sC1[(row0 + r) * SC1 + 64 + col] = (_Float16)acch[r];
        }
      }
      __syncthreads();  // C1 complete

      // Stage 2: gate = C1 (64x128) @ [Ww2|Wu2]^T  -> K = 128, 4 WMMAs
      v8f acc = {};
      const _Float16* B2 = sW2 + g * 64 * SW2;
      for (int kc = 0; kc < 128; kc += 32) {
        v16h ac = load_fragA(sC1, ti * 16 + idx, SC1, kc, hsel);
        v16h bw = load_fragB(B2,  tj * 16 + idx, SW2, kc, hsel);
        acc = wmma_f16(ac, bw, acc);
      }
      __syncthreads();  // C1 reads done before next gate overwrites
      gate[g] = acc;
    }

    // Elementwise LSTM update, entirely in registers (C-layout fragments)
    {
      int row0 = ti * 16 + hsel * 8;
      int col  = tj * 16 + idx;
      for (int r = 0; r < 8; ++r) {
        float z  = sigmoidf(gate[0][r]);
        float rr = sigmoidf(gate[1][r]);
        float o  = sigmoidf(gate[2][r]);
        float cv = rr * c_acc[r] + z * rr;
        float hv = o * sigmoidf(cv);
        c_acc[r] = cv;
        h_acc[r] = hv;
        outs[(size_t)t * 4096 + (row0 + r) * 64 + col] = hv;   // stream h_t out
        sHt[col * SXT + (row0 + r)] = (_Float16)hv;            // transposed image for next step
      }
    }
    // No barrier needed here: Ht/Xt re-reads are fenced by the barrier at loop top.
  }

  // h_last, c_last
  {
    float* hL = out + (size_t)64 * 128 * 4096;            // 33,554,432
    float* cL = hL + (size_t)64 * 4096;                   // + 262,144
    int row0 = ti * 16 + hsel * 8;
    int col  = tj * 16 + idx;
    for (int r = 0; r < 8; ++r) {
      hL[(size_t)n * 4096 + (row0 + r) * 64 + col] = h_acc[r];
      cL[(size_t)n * 4096 + (row0 + r) * 64 + col] = c_acc[r];
    }
  }
}

extern "C" void kernel_launch(void* const* d_in, const int* in_sizes, int n_in,
                              void* d_out, int out_size, void* d_ws, size_t ws_size,
                              hipStream_t stream) {
  (void)in_sizes; (void)n_in; (void)out_size; (void)d_ws; (void)ws_size;
  const float* inputs = (const float*)d_in[0];
  const float* Ww1    = (const float*)d_in[1];
  const float* Ww2    = (const float*)d_in[2];
  const float* Wu1    = (const float*)d_in[3];
  const float* Wu2    = (const float*)d_in[4];
  float* out = (float*)d_out;

  const size_t shmem = (size_t)SMEM_HALVES * 2;  // 143,360 bytes (< 320 KB WGP LDS)
  lstm_tl_kernel<<<dim3(64), dim3(512), shmem, stream>>>(inputs, Ww1, Ww2, Wu1, Wu2, out);
}
